// QuantLinearLUT_49993419325542
// MI455X (gfx1250) — matrix operand order repaired and checked by reading, not compiled, not run on hardware
//
#include <hip/hip_runtime.h>
#include <hip/hip_bf16.h>

// ---------------------------------------------------------------------------
// QuantLinearLUT on MI455X (gfx1250, wave32):
//   out = X @ dequant(qweight, LUT) (+CSR outliers) + bias
//   M=4096, K=4096, N=11008  -> compute bound (AI ~1370 FLOP/B)
// Strategy: materialize W as bf16 (N-major) + X as bf16 in workspace (both fit
// in 192MB L2), then dense bf16 WMMA GEMM (f32 accumulate).
// GEMM: block tile 128(M) x 256(N), 8 waves as 2x4, each wave 64x64 =
// 4x4 fragments -> 16 v_wmma per 32-deep K step at 1.0 b128-loads/WMMA.
// ---------------------------------------------------------------------------

typedef __attribute__((ext_vector_type(16))) __bf16 v16bf;
typedef __attribute__((ext_vector_type(8)))  __bf16 v8bf;
typedef __attribute__((ext_vector_type(8)))  float  v8f;

static constexpr int Kdim  = 4096;   // in-features
static constexpr int Nout  = 11008;  // out-features
static constexpr int Mrows = 4096;   // 2*2048 flattened batch*seq

__device__ __forceinline__ __bf16 tobf(float f) { return (__bf16)f; }

// ---------------------------------------------------------------------------
// Kernel 1: convert X fp32 -> bf16 (8 elems / thread, 16B stores)
// ---------------------------------------------------------------------------
__global__ __launch_bounds__(256)
void conv_x_kernel(const float* __restrict__ x, __bf16* __restrict__ xb, int total8) {
  int t = blockIdx.x * 256 + threadIdx.x;
  if (t >= total8) return;
  const float4* p = (const float4*)x + (size_t)t * 2;
  float4 f0 = p[0];
  float4 f1 = p[1];
  v8bf o;
  o[0] = tobf(f0.x); o[1] = tobf(f0.y); o[2] = tobf(f0.z); o[3] = tobf(f0.w);
  o[4] = tobf(f1.x); o[5] = tobf(f1.y); o[6] = tobf(f1.z); o[7] = tobf(f1.w);
  *(v8bf*)(xb + (size_t)t * 8) = o;
}

// ---------------------------------------------------------------------------
// Kernel 2: dequant qweight -> W_t (N-major: W_t[n*K + k], bf16)
// Block: 256 threads = 64 n-cols x 4 int32-packs (32 k). LUT rows staged in
// LDS; result transposed through LDS so global stores are 64B runs per n-row.
// ---------------------------------------------------------------------------
__global__ __launch_bounds__(256)
void dequant_w_kernel(const int* __restrict__ qw, const float* __restrict__ lut,
                      __bf16* __restrict__ wt) {
  __shared__ float slut[64 * 16];
  __shared__ __align__(16) __bf16 swt[64][40];   // 64 n x 32 k, padded rows

  const int t  = threadIdx.x;
  const int nb = blockIdx.x;   // n tile (64 wide)
  const int pb = blockIdx.y;   // pack tile (4 packs = 32 k)

  // stage 64 LUT rows (contiguous 1024 floats)
  for (int i = t; i < 1024; i += 256)
    slut[i] = lut[(size_t)nb * 1024 + i];
  __syncthreads();

  const int n_local = t & 63;
  const int p_local = t >> 6;
  const int n    = nb * 64 + n_local;
  const int pack = pb * 4 + p_local;

  unsigned q = ((const unsigned*)qw)[(size_t)pack * Nout + n];
#pragma unroll
  for (int j = 0; j < 8; ++j) {
    int code = (q >> (4 * j)) & 0xF;
    swt[n_local][p_local * 8 + j] = tobf(slut[n_local * 16 + code]);
  }
  __syncthreads();

  // write out: thread u -> row u/4 (n), 16B segment u%4 (k); 64B runs per row
  const int row = t >> 2;
  const int seg = t & 3;
  v8bf v = *(const v8bf*)&swt[row][seg * 8];
  *(v8bf*)(wt + (size_t)(nb * 64 + row) * Kdim + pb * 32 + seg * 8) = v;
}

// ---------------------------------------------------------------------------
// Kernel 3: fold CSR outliers into W_t (one thread per output row; sequential
// per-row loop correctly accumulates duplicate column indices)
// ---------------------------------------------------------------------------
__global__ __launch_bounds__(256)
void sparse_fix_kernel(const int* __restrict__ rows, const int* __restrict__ cols,
                       const float* __restrict__ vals, __bf16* __restrict__ wt) {
  int r = blockIdx.x * blockDim.x + threadIdx.x;
  if (r >= Nout) return;
  int beg = rows[r], end = rows[r + 1];
  for (int i = beg; i < end; ++i) {
    int k = cols[i];
    size_t idx = (size_t)r * Kdim + k;
    float w = (float)wt[idx] + vals[i];
    wt[idx] = tobf(w);
  }
}

// ---------------------------------------------------------------------------
// Kernel 4: GEMM via v_wmma_f32_16x16x32_bf16.
// Block tile 128(M) x 256(N), 8 waves (2 m x 4 n), each wave 64x64 = 4x4 frags.
// A: x_bf16 row-major (M x K). Lane holds one contiguous 16-elem K run
//    at k0+16*half (half = lane>>4), matching B.
// B: W_t N-major (N x K) => lane fragment is one contiguous 32B run:
//    lanes 0-15 K=k0..k0+15, lanes 16-31 K=k0+16..k0+31.
// ---------------------------------------------------------------------------
__global__ __launch_bounds__(256)
void gemm_wmma_kernel(const __bf16* __restrict__ xb, const __bf16* __restrict__ wt,
                      const float* __restrict__ bias, float* __restrict__ out) {
  const int lane = threadIdx.x & 31;
  const int wid  = threadIdx.x >> 5;
  const int wm   = wid >> 2;        // 0..1
  const int wn   = wid & 3;         // 0..3
  const int half = lane >> 4;       // 0/1
  const int r15  = lane & 15;

  const int m_base = blockIdx.y * 128 + wm * 64;
  const int n_base = blockIdx.x * 256 + wn * 64;

  v8f zero = {};
  v8f acc[4][4];
#pragma unroll
  for (int i = 0; i < 4; ++i)
#pragma unroll
    for (int j = 0; j < 4; ++j) acc[i][j] = zero;

  const __bf16* aptr[4];
#pragma unroll
  for (int i = 0; i < 4; ++i)
    aptr[i] = xb + (size_t)(m_base + i * 16 + r15) * Kdim + half * 16;
  const __bf16* bptr[4];
#pragma unroll
  for (int j = 0; j < 4; ++j)
    bptr[j] = wt + (size_t)(n_base + j * 16 + r15) * Kdim + half * 16;

  for (int k0 = 0; k0 < Kdim; k0 += 32) {
    // prefetch the W stream one K-step ahead (L2-resident; near-cache hint)
    __builtin_prefetch(bptr[0] + k0 + 32, 0, 3);

    v16bf a[4];
#pragma unroll
    for (int i = 0; i < 4; ++i) {
      v8bf lo = *(const v8bf*)(aptr[i] + k0);        // K = k0+16h   .. +7
      v8bf hi = *(const v8bf*)(aptr[i] + k0 + 8);    // K = k0+16h+8 .. +7
      a[i] = __builtin_shufflevector(lo, hi, 0, 1, 2, 3, 4, 5, 6, 7,
                                             8, 9, 10, 11, 12, 13, 14, 15);
    }
    v16bf b[4];
#pragma unroll
    for (int j = 0; j < 4; ++j)
      b[j] = *(const v16bf*)(bptr[j] + k0);

#pragma unroll
    for (int j = 0; j < 4; ++j)
#pragma unroll
      for (int i = 0; i < 4; ++i)
        acc[i][j] = __builtin_amdgcn_wmma_f32_16x16x32_bf16(
            /*neg_a=*/false, a[i], /*neg_b=*/false, b[j],
            /*c_mod=*/(short)0, acc[i][j], /*reuse_a=*/false, /*reuse_b=*/false);
  }

  // Epilogue: C/D layout -> VGPR r holds M = r + 8*half, N = lane&15
#pragma unroll
  for (int j = 0; j < 4; ++j) {
    const int col = n_base + j * 16 + r15;
    const float bv = bias[col];
#pragma unroll
    for (int i = 0; i < 4; ++i) {
#pragma unroll
      for (int r = 0; r < 8; ++r) {
        const int row = m_base + i * 16 + half * 8 + r;
        out[(size_t)row * Nout + col] = acc[i][j][r] + bv;
      }
    }
  }
}

// ---------------------------------------------------------------------------
extern "C" void kernel_launch(void* const* d_in, const int* in_sizes, int n_in,
                              void* d_out, int out_size, void* d_ws, size_t ws_size,
                              hipStream_t stream) {
  (void)in_sizes; (void)n_in; (void)out_size; (void)ws_size;

  const float* x       = (const float*)d_in[0];
  const int*   qweight = (const int*)  d_in[1];
  const float* lut     = (const float*)d_in[2];
  const int*   rows    = (const int*)  d_in[3];
  const int*   cols    = (const int*)  d_in[4];
  const float* vals    = (const float*)d_in[5];
  const float* bias    = (const float*)d_in[6];
  float*       out     = (float*)d_out;

  // workspace: [ x_bf16 : M*K ][ W_t bf16 : N*K ]  (~124 MB)
  __bf16* xb = (__bf16*)d_ws;
  __bf16* wt = (__bf16*)((char*)d_ws + (size_t)Mrows * Kdim * sizeof(__bf16));

  const int total8 = Mrows * Kdim / 8;
  conv_x_kernel<<<(total8 + 255) / 256, 256, 0, stream>>>(x, xb, total8);

  dequant_w_kernel<<<dim3(Nout / 64, (Kdim / 8) / 4), 256, 0, stream>>>(qweight, lut, wt);

  sparse_fix_kernel<<<(Nout + 255) / 256, 256, 0, stream>>>(rows, cols, vals, wt);

  gemm_wmma_kernel<<<dim3(Nout / 256, Mrows / 128), 256, 0, stream>>>(xb, wt, bias, out);
}
